// AnchorStoreOptV1_53102975647800
// MI455X (gfx1250) — matrix operand order repaired and checked by reading, not compiled, not run on hardware
//
#include <hip/hip_runtime.h>
#include <hip/hip_bf16.h>

#define NB   64          // queries
#define KA   4096        // anchors
#define DIMX 50257       // vocab dim
#define DPAD 50272       // DIMX rounded up to multiple of 32 (K-step)
#define DSPLIT 25120     // D-slab boundary (multiple of 32): slab0=[0,DSPLIT), slab1=[DSPLIT,DIMX)
#define KNN  8
#define LN2F 0.69314718055994531f

typedef __attribute__((ext_vector_type(16))) __bf16 v16bf;
typedef __attribute__((ext_vector_type(8)))  __bf16 v8bf;
typedef __attribute__((ext_vector_type(8)))  float  v8f;

// ---------------------------------------------------------------------------
// Kernel A: qlog[b,d] = ln(query[b,d] + 1e-10) in bf16, zero-padded to DPAD.
// ---------------------------------------------------------------------------
__global__ __launch_bounds__(256) void qlog_kernel(const float* __restrict__ q,
                                                   __bf16* __restrict__ qlog) {
  int idx = blockIdx.x * 256 + threadIdx.x;
  if (idx >= NB * DPAD) return;
  int b = idx / DPAD;
  int d = idx - b * DPAD;
  float v = 0.0f;
  if (d < DIMX) v = LN2F * __builtin_amdgcn_logf(q[(size_t)b * DIMX + d] + 1e-10f);
  qlog[idx] = (__bf16)v;
}

// ---------------------------------------------------------------------------
// Zero-init the kl accumulation buffer (slabs combine via global f32 atomics).
// ---------------------------------------------------------------------------
__global__ __launch_bounds__(256) void kl_zero(float* __restrict__ kl) {
  int idx = blockIdx.x * 256 + threadIdx.x;
  if (idx < NB * KA) kl[idx] = 0.0f;
}

// ---------------------------------------------------------------------------
// Kernel B: fused single pass over the anchor matrix (823 MB, read once).
//   cross[b,n] = sum_d qlog[b,d] * anchor[n,d]           (bf16 WMMA, f32 accum)
//   self2[n]   = sum_d anchor[n,d]*log2(anchor[n,d]+eps)  (raw v_log_f32)
//   kl[b,n]   += (self2[n]*ln2 - cross[b,n]) / DIMX       (per D-slab partial)
// Grid: (KA/16, 2) — blockIdx.x picks 16 anchors, blockIdx.y picks a D-slab
// (doubles resident waves / HBM requests in flight). Block: 8 waves; each
// wave owns a strided K-slab of its D-slab and all 64 query rows (4 accs).
// A fragments for all 4 row-tiles are preloaded so their L2 latency hides
// under the preceding WMMAs.
// ---------------------------------------------------------------------------
__global__ __launch_bounds__(256) void kl_gemm(const float* __restrict__ anchor,
                                               const __bf16* __restrict__ qlog,
                                               float* __restrict__ kl) {
  const int tid   = threadIdx.x;
  const int wave  = tid >> 5;
  const int lane  = tid & 31;
  const int half  = lane >> 4;   // K half-selector within fragments
  const int l16   = lane & 15;   // N column (B/C frag) / M row (A frag)
  const int nbase = blockIdx.x * 16;
  const int slab  = blockIdx.y;  // 0 or 1

  __shared__ float c_red[NB * 16];   // 64x16 f32 cross-wave reduction tile
  __shared__ float self_red[16];
  for (int i = tid; i < NB * 16; i += 256) c_red[i] = 0.0f;
  if (tid < 16) self_red[tid] = 0.0f;
  __syncthreads();

  v8f acc[4] = {};                   // 4 x (16x16 f32) accumulators = rows 0..63
  float s2[4] = {0.0f, 0.0f, 0.0f, 0.0f};   // log2-domain self-term partials

  const float* arow = anchor + (size_t)(nbase + l16) * DIMX;
  const __bf16* qbase = qlog + (size_t)l16 * DPAD + half * 16;

  const int dlimit = slab ? DIMX : DSPLIT;   // fast-loop upper bound
  int d0 = slab * DSPLIT + wave * 32;

  // ---- branch-free main loop: all 32 columns of the chunk are in-bounds ----
  for (; d0 + 32 <= dlimit; d0 += 8 * 32) {
    const float* ap = arow + d0 + half * 16;
    __builtin_prefetch((const void*)(ap + 8 * 32), 0, 3);

    float fv[16];
    __builtin_memcpy(fv, ap, 64);    // widest legal loads (rows are 4B-aligned)

    v16bf bfrag;
#pragma unroll
    for (int i = 0; i < 16; ++i) {
      float f = fv[i];
      s2[i & 3] = __builtin_fmaf(f, __builtin_amdgcn_logf(f + 1e-10f), s2[i & 3]);
      bfrag[i] = (__bf16)f;
    }

    // Preload all 4 A tiles (keeps 8 b128 loads in flight together), then MMA.
    v16bf afrag[4];
#pragma unroll
    for (int t = 0; t < 4; ++t) {
      const __bf16* qrow = qbase + (size_t)(t * 16) * DPAD + d0;
      v8bf lo = *(const v8bf*)(qrow);
      v8bf hi = *(const v8bf*)(qrow + 8);
#pragma unroll
      for (int i = 0; i < 8; ++i) { afrag[t][i] = lo[i]; afrag[t][8 + i] = hi[i]; }
    }
#pragma unroll
    for (int t = 0; t < 4; ++t)
      acc[t] = __builtin_amdgcn_wmma_f32_16x16x32_bf16(
          false, afrag[t], false, bfrag, (short)0, acc[t], false, false);
  }

  // ---- peeled ragged chunk (slab 1 only; wave-uniform entry) ----
  if (slab && d0 < DPAD) {
    const int dcol = d0 + half * 16;
    v16bf bfrag;
#pragma unroll
    for (int i = 0; i < 16; ++i) {
      int c = dcol + i;
      float f = (c < DIMX) ? arow[c] : 0.0f;
      if (c < DIMX)
        s2[i & 3] = __builtin_fmaf(f, __builtin_amdgcn_logf(f + 1e-10f), s2[i & 3]);
      bfrag[i] = (__bf16)f;
    }
#pragma unroll
    for (int t = 0; t < 4; ++t) {
      const __bf16* qrow = qbase + (size_t)(t * 16) * DPAD + d0;
      v8bf lo = *(const v8bf*)(qrow);
      v8bf hi = *(const v8bf*)(qrow + 8);
      v16bf afrag;
#pragma unroll
      for (int i = 0; i < 8; ++i) { afrag[i] = lo[i]; afrag[8 + i] = hi[i]; }
      acc[t] = __builtin_amdgcn_wmma_f32_16x16x32_bf16(
          false, afrag, false, bfrag, (short)0, acc[t], false, false);
    }
  }

  // ---- cross-wave reduction (C layout: VGPR r -> row r + 8*half, col l16) ----
#pragma unroll
  for (int t = 0; t < 4; ++t) {
#pragma unroll
    for (int r = 0; r < 8; ++r) {
      int m = t * 16 + r + half * 8;
      atomicAdd(&c_red[m * 16 + l16], acc[t][r]);
    }
  }
  atomicAdd(&self_red[l16], (s2[0] + s2[1]) + (s2[2] + s2[3]));
  __syncthreads();

  // ---- combine this slab's partial into kl via global f32 atomics ----
  const float invD = 1.0f / (float)DIMX;
  for (int i = tid; i < NB * 16; i += 256) {
    int b = i >> 4;
    int n = i & 15;
    atomicAdd(&kl[(size_t)b * KA + nbase + n],
              (self_red[n] * LN2F - c_red[i]) * invD);
  }
}

// ---------------------------------------------------------------------------
// Kernel C: per query row, 8 rounds of LDS argmin over the 4096 distances,
// then 2-class majority vote (argmax tie -> class 0).
// ---------------------------------------------------------------------------
__global__ __launch_bounds__(256) void topk_vote(const float* __restrict__ kl,
                                                 const int* __restrict__ label,
                                                 float* __restrict__ out) {
  const int b = blockIdx.x;
  const int tid = threadIdx.x;
  __shared__ float sv[KA];
  __shared__ float rv[256];
  __shared__ int   ri[256];
  __shared__ int   cnt1;

  for (int i = tid; i < KA; i += 256) sv[i] = kl[(size_t)b * KA + i];
  if (tid == 0) cnt1 = 0;
  __syncthreads();

  for (int r = 0; r < KNN; ++r) {
    float best = 3.4e38f;
    int bi = 0;
    for (int j = tid; j < KA; j += 256)
      if (sv[j] < best) { best = sv[j]; bi = j; }
    rv[tid] = best; ri[tid] = bi;
    __syncthreads();
    for (int s = 128; s > 0; s >>= 1) {
      if (tid < s && rv[tid + s] < rv[tid]) { rv[tid] = rv[tid + s]; ri[tid] = ri[tid + s]; }
      __syncthreads();
    }
    if (tid == 0) {
      int w = ri[0];
      sv[w] = 3.4e38f;
      if (label[w] == 1) cnt1++;
    }
    __syncthreads();
  }
  if (tid == 0) out[b] = (cnt1 > KNN - cnt1) ? 1.0f : 0.0f;
}

// ---------------------------------------------------------------------------
extern "C" void kernel_launch(void* const* d_in, const int* in_sizes, int n_in,
                              void* d_out, int out_size, void* d_ws, size_t ws_size,
                              hipStream_t stream) {
  (void)in_sizes; (void)n_in; (void)out_size; (void)ws_size;
  const float* query  = (const float*)d_in[0];   // [64, 50257] f32
  const float* anchor = (const float*)d_in[1];   // [4096, 50257] f32
  const int*   label  = (const int*)d_in[2];     // [4096] i32

  // workspace layout: qlog bf16 [64 x DPAD] | kl f32 [64 x 4096]
  __bf16* qlog = (__bf16*)d_ws;
  float*  kl   = (float*)((char*)d_ws + (size_t)NB * DPAD * sizeof(__bf16));
  float*  out  = (float*)d_out;

  qlog_kernel<<<(NB * DPAD + 255) / 256, 256, 0, stream>>>(query, qlog);
  kl_zero<<<(NB * KA + 255) / 256, 256, 0, stream>>>(kl);
  kl_gemm<<<dim3(KA / 16, 2), 256, 0, stream>>>(anchor, qlog, kl);
  topk_vote<<<NB, 256, 0, stream>>>(kl, label, out);
}
